// GraphNet_14577119003008
// MI455X (gfx1250) — matrix operand order repaired and checked by reading, not compiled, not run on hardware
//
#include <hip/hip_runtime.h>
#include <hip/hip_bf16.h>

typedef __attribute__((ext_vector_type(16))) _Float16 v16h;
typedef __attribute__((ext_vector_type(8)))  float    v8f;

#define NW 4   // waves per block (wave32)
#define EPS 1e-5f

// ---------- helpers ----------

__device__ inline void cvt_copy(const float* __restrict__ s, _Float16* d, int n) {
    // n multiple of 4; source 16B aligned
    for (int i = 0; i < n; i += 4) {
        float4 v = *(const float4*)(s + i);
        d[i + 0] = (_Float16)v.x;
        d[i + 1] = (_Float16)v.y;
        d[i + 2] = (_Float16)v.z;
        d[i + 3] = (_Float16)v.w;
    }
}

// A fragment (16x32 f16) from row-major f16 LDS tile [16][ldk]
// Layout: lane m=lane&15 (row M), kh=lane>>4; halves j -> K = kh*16 + j
__device__ inline v16h load_a(const _Float16* sf, int ldk, int kbase, int lane) {
    int m = lane & 15, kh = lane >> 4;
    const _Float16* p = sf + m * ldk + kbase + kh * 16;
    v16h a;
#pragma unroll
    for (int j = 0; j < 8; ++j) { a[j] = p[j]; a[8 + j] = p[8 + j]; }
    return a;
}

// B fragment (32x16 f16) from transposed weights WT[N][ldk] (f16, 32B-aligned rows)
// lane n=lane&15 (col N), kh=lane>>4; halves j -> K = kh*16 + j
__device__ inline v16h load_bT(const _Float16* WT, int ldk, int kbase, int nbase, int lane) {
    int n = (lane & 15) + nbase, kh = lane >> 4;
    return *(const v16h*)(WT + (size_t)n * ldk + kbase + kh * 16);
}

__device__ inline v8f wmma_f16(v16h a, v16h b, v8f c) {
    return __builtin_amdgcn_wmma_f32_16x16x32_f16(false, a, false, b, (short)0, c, false, false);
}

// 2-layer MLP on a 16-row tile:  [16,KPAD] -> relu -> [16,64] -> relu -> of[16,NOUT] (f32)
template <int KPAD, int NOUT>
__device__ inline void mlp2(const _Float16* sf, _Float16* hf, float* of,
                            const _Float16* __restrict__ WT1, const float* __restrict__ b1,
                            const _Float16* __restrict__ WT2, const float* __restrict__ b2,
                            int lane) {
    int m = lane & 15, kh = lane >> 4;
    // layer 1: K = KPAD, N = 64 (4 tiles of 16)
#pragma unroll
    for (int t = 0; t < 4; ++t) {
        v8f cc = {};
#pragma unroll
        for (int ch = 0; ch < KPAD / 32; ++ch) {
            v16h a = load_a(sf, KPAD, ch * 32, lane);
            v16h b = load_bT(WT1, KPAD, ch * 32, t * 16, lane);
            cc = wmma_f16(a, b, cc);
        }
        float bias = b1[t * 16 + m];
#pragma unroll
        for (int r = 0; r < 8; ++r) {
            int M = kh * 8 + r;
            float v = cc[r] + bias;
            v = v > 0.f ? v : 0.f;
            hf[M * 64 + t * 16 + m] = (_Float16)v;
        }
    }
    __builtin_amdgcn_wave_barrier();
    // layer 2: K = 64, N = NOUT
#pragma unroll
    for (int t = 0; t < NOUT / 16; ++t) {
        v8f cc = {};
#pragma unroll
        for (int ch = 0; ch < 2; ++ch) {
            v16h a = load_a(hf, 64, ch * 32, lane);
            v16h b = load_bT(WT2, 64, ch * 32, t * 16, lane);
            cc = wmma_f16(a, b, cc);
        }
        float bias = b2[t * 16 + m];
#pragma unroll
        for (int r = 0; r < 8; ++r) {
            int M = kh * 8 + r;
            float v = cc[r] + bias;
            v = v > 0.f ? v : 0.f;
            of[M * NOUT + t * 16 + m] = v;
        }
    }
    __builtin_amdgcn_wave_barrier();
}

// ---------- prep kernels ----------

__global__ void k_zero(float* p, int n) {
    int i = blockIdx.x * blockDim.x + threadIdx.x;
    if (i < n) p[i] = 0.f;
}

// dst[n*KPAD + k] = (k < K) ? src[k*Nmat + n] : 0   (f32 -> f16, transpose + pad)
__global__ void k_tpad(const float* __restrict__ src, _Float16* dst, int K, int Nmat, int KPAD) {
    int i = blockIdx.x * blockDim.x + threadIdx.x;
    if (i >= Nmat * KPAD) return;
    int n = i / KPAD, k = i % KPAD;
    dst[(size_t)n * KPAD + k] = (k < K) ? (_Float16)src[(size_t)k * Nmat + n] : (_Float16)0.f;
}

// ---------- edge model ----------
// e_cat = [x[row] | x[col] | edge_attr | u[e_idx]] (176 -> pad 192) -> MLP -> LN
// fused: store edge_out, atomicAdd into agg_e[col], gsum_e[g], gcnt_e[g]
__global__ __launch_bounds__(NW * 32) void k_edge(
    const float* __restrict__ x, const float* __restrict__ ea, const float* __restrict__ u,
    const int* __restrict__ eidx, const int* __restrict__ egid,
    const _Float16* __restrict__ WT1, const float* __restrict__ b1,
    const _Float16* __restrict__ WT2, const float* __restrict__ b2,
    const float* __restrict__ gam, const float* __restrict__ bet,
    float* __restrict__ edge_out, float* agg_e, float* gsum_e, float* gcnt_e, int E) {
    __shared__ _Float16 s_sf[NW][16 * 192];
    __shared__ _Float16 s_hf[NW][16 * 64];
    __shared__ float    s_of[NW][16 * 48];
    int wave = threadIdx.x >> 5, lane = threadIdx.x & 31;
    int tile = blockIdx.x * NW + wave;
    if (tile * 16 >= E) return;              // wave-uniform
    int m = lane & 15, kh = lane >> 4;
    int e = tile * 16 + m;
    int r = eidx[e], c = eidx[E + e], g = egid[e];
    _Float16* row = &s_sf[wave][m * 192];
    if (kh == 0) {
        cvt_copy(x + (size_t)r * 48, row, 48);
        cvt_copy(x + (size_t)c * 48, row + 48, 48);
    } else {
        cvt_copy(ea + (size_t)e * 48, row + 96, 48);
        cvt_copy(u + (size_t)g * 32, row + 144, 32);
        for (int k = 0; k < 16; ++k) row[176 + k] = (_Float16)0.f;
    }
    __builtin_amdgcn_wave_barrier();
    mlp2<192, 48>(s_sf[wave], s_hf[wave], s_of[wave], WT1, b1, WT2, b2, lane);
    // LayerNorm over 48 + fused scatter
    const float* orow = &s_of[wave][m * 48];
    float mu = 0.f;
#pragma unroll
    for (int k = 0; k < 48; ++k) mu += orow[k];
    mu *= (1.f / 48.f);
    float var = 0.f;
#pragma unroll
    for (int k = 0; k < 48; ++k) { float d = orow[k] - mu; var += d * d; }
    var *= (1.f / 48.f);
    float rs = rsqrtf(var + EPS);
    int k0 = kh * 24;
    for (int k = k0; k < k0 + 24; ++k) {
        float y = (orow[k] - mu) * rs * gam[k] + bet[k];
        edge_out[(size_t)e * 48 + k] = y;
        atomicAdd(&agg_e[(size_t)c * 48 + k], y);
        atomicAdd(&gsum_e[(size_t)g * 48 + k], y);
    }
    if (kh == 0) atomicAdd(&gcnt_e[g], 1.0f);
}

// ---------- node model ----------
// x_cat = [x | agg_e | u[v_idx]] (128) -> MLP -> LN ; fused gsum_v/gcnt_v scatter
__global__ __launch_bounds__(NW * 32) void k_node(
    const float* __restrict__ x, const float* __restrict__ agg_e, const float* __restrict__ u,
    const int* __restrict__ vgid,
    const _Float16* __restrict__ WT1, const float* __restrict__ b1,
    const _Float16* __restrict__ WT2, const float* __restrict__ b2,
    const float* __restrict__ gam, const float* __restrict__ bet,
    float* __restrict__ x_out, float* gsum_v, float* gcnt_v, int N) {
    __shared__ _Float16 s_sf[NW][16 * 128];
    __shared__ _Float16 s_hf[NW][16 * 64];
    __shared__ float    s_of[NW][16 * 48];
    int wave = threadIdx.x >> 5, lane = threadIdx.x & 31;
    int tile = blockIdx.x * NW + wave;
    if (tile * 16 >= N) return;
    int m = lane & 15, kh = lane >> 4;
    int n = tile * 16 + m;
    int g = vgid[n];
    _Float16* row = &s_sf[wave][m * 128];
    if (kh == 0) {
        cvt_copy(x + (size_t)n * 48, row, 48);
    } else {
        cvt_copy(agg_e + (size_t)n * 48, row + 48, 48);
        cvt_copy(u + (size_t)g * 32, row + 96, 32);
    }
    __builtin_amdgcn_wave_barrier();
    mlp2<128, 48>(s_sf[wave], s_hf[wave], s_of[wave], WT1, b1, WT2, b2, lane);
    const float* orow = &s_of[wave][m * 48];
    float mu = 0.f;
#pragma unroll
    for (int k = 0; k < 48; ++k) mu += orow[k];
    mu *= (1.f / 48.f);
    float var = 0.f;
#pragma unroll
    for (int k = 0; k < 48; ++k) { float d = orow[k] - mu; var += d * d; }
    var *= (1.f / 48.f);
    float rs = rsqrtf(var + EPS);
    int k0 = kh * 24;
    for (int k = k0; k < k0 + 24; ++k) {
        float y = (orow[k] - mu) * rs * gam[k] + bet[k];
        x_out[(size_t)n * 48 + k] = y;
        atomicAdd(&gsum_v[(size_t)g * 48 + k], y);
    }
    if (kh == 0) atomicAdd(&gcnt_v[g], 1.0f);
}

// ---------- global model ----------
// u_cat = [u | mean_v | mean_e] (128) -> MLP -> LN(32)
__global__ __launch_bounds__(NW * 32) void k_global(
    const float* __restrict__ u,
    const float* __restrict__ gsum_v, const float* __restrict__ gcnt_v,
    const float* __restrict__ gsum_e, const float* __restrict__ gcnt_e,
    const _Float16* __restrict__ WT1, const float* __restrict__ b1,
    const _Float16* __restrict__ WT2, const float* __restrict__ b2,
    const float* __restrict__ gam, const float* __restrict__ bet,
    float* __restrict__ u_out, int G) {
    __shared__ _Float16 s_sf[NW][16 * 128];
    __shared__ _Float16 s_hf[NW][16 * 64];
    __shared__ float    s_of[NW][16 * 32];
    int wave = threadIdx.x >> 5, lane = threadIdx.x & 31;
    int tile = blockIdx.x * NW + wave;
    if (tile * 16 >= G) return;
    int m = lane & 15, kh = lane >> 4;
    int g = tile * 16 + m;
    _Float16* row = &s_sf[wave][m * 128];
    if (kh == 0) {
        cvt_copy(u + (size_t)g * 32, row, 32);
        float inv = 1.f / fmaxf(gcnt_v[g], 1.f);
        for (int k = 0; k < 48; ++k) row[32 + k] = (_Float16)(gsum_v[(size_t)g * 48 + k] * inv);
    } else {
        float inv = 1.f / fmaxf(gcnt_e[g], 1.f);
        for (int k = 0; k < 48; ++k) row[80 + k] = (_Float16)(gsum_e[(size_t)g * 48 + k] * inv);
    }
    __builtin_amdgcn_wave_barrier();
    mlp2<128, 32>(s_sf[wave], s_hf[wave], s_of[wave], WT1, b1, WT2, b2, lane);
    const float* orow = &s_of[wave][m * 32];
    float mu = 0.f;
#pragma unroll
    for (int k = 0; k < 32; ++k) mu += orow[k];
    mu *= (1.f / 32.f);
    float var = 0.f;
#pragma unroll
    for (int k = 0; k < 32; ++k) { float d = orow[k] - mu; var += d * d; }
    var *= (1.f / 32.f);
    float rs = rsqrtf(var + EPS);
    int k0 = kh * 16;
    for (int k = k0; k < k0 + 16; ++k)
        u_out[(size_t)g * 32 + k] = (orow[k] - mu) * rs * gam[k] + bet[k];
}

// ---------- host ----------

extern "C" void kernel_launch(void* const* d_in, const int* in_sizes, int n_in,
                              void* d_out, int out_size, void* d_ws, size_t ws_size,
                              hipStream_t stream) {
    const float* x    = (const float*)d_in[0];
    const float* ea   = (const float*)d_in[1];
    const float* u    = (const float*)d_in[2];
    const int*   eidx = (const int*)d_in[3];
    const int*   vgid = (const int*)d_in[4];
    const int*   egid = (const int*)d_in[5];
    const float* e_w1 = (const float*)d_in[6];  const float* e_b1 = (const float*)d_in[7];
    const float* e_w2 = (const float*)d_in[8];  const float* e_b2 = (const float*)d_in[9];
    const float* e_g  = (const float*)d_in[10]; const float* e_be = (const float*)d_in[11];
    const float* v_w1 = (const float*)d_in[12]; const float* v_b1 = (const float*)d_in[13];
    const float* v_w2 = (const float*)d_in[14]; const float* v_b2 = (const float*)d_in[15];
    const float* v_g  = (const float*)d_in[16]; const float* v_be = (const float*)d_in[17];
    const float* u_w1 = (const float*)d_in[18]; const float* u_b1 = (const float*)d_in[19];
    const float* u_w2 = (const float*)d_in[20]; const float* u_b2 = (const float*)d_in[21];
    const float* u_g  = (const float*)d_in[22]; const float* u_be = (const float*)d_in[23];

    const int N = in_sizes[0] / 48;
    const int E = in_sizes[1] / 48;
    const int G = in_sizes[2] / 32;

    // workspace layout (256B-aligned regions)
    char* ws = (char*)d_ws;
    _Float16* wt_e1 = (_Float16*)(ws + 0);                  // [64][192]  24576 B
    _Float16* wt_e2 = (_Float16*)(ws + 24576);              // [48][64]    6144 B
    _Float16* wt_v1 = (_Float16*)(ws + 30720);              // [64][128]  16384 B
    _Float16* wt_v2 = (_Float16*)(ws + 47104);              // [48][64]    6144 B
    _Float16* wt_u1 = (_Float16*)(ws + 53248);              // [64][128]  16384 B
    _Float16* wt_u2 = (_Float16*)(ws + 69632);              // [32][64]    4096 B
    size_t o = 73728;
    float* agg_e  = (float*)(ws + o);  o += (size_t)N * 48 * 4;
    float* gsum_v = (float*)(ws + o);  o += (size_t)G * 48 * 4;
    float* gsum_e = (float*)(ws + o);  o += (size_t)G * 48 * 4;
    float* gcnt_v = (float*)(ws + o);  o += (size_t)G * 4;
    float* gcnt_e = (float*)(ws + o);  o += (size_t)G * 4;

    float* x_out    = (float*)d_out;
    float* edge_out = x_out + (size_t)N * 48;
    float* u_out    = edge_out + (size_t)E * 48;

    // 1) zero accumulators (every call: workspace not re-poisoned between replays)
    int zn = N * 48 + G * 48 * 2 + G * 2;
    k_zero<<<dim3((zn + 255) / 256), dim3(256), 0, stream>>>(agg_e, zn);

    // 2) transpose + pad weights to f16
    k_tpad<<<dim3((64 * 192 + 255) / 256), dim3(256), 0, stream>>>(e_w1, wt_e1, 176, 64, 192);
    k_tpad<<<dim3((48 * 64  + 255) / 256), dim3(256), 0, stream>>>(e_w2, wt_e2, 64, 48, 64);
    k_tpad<<<dim3((64 * 128 + 255) / 256), dim3(256), 0, stream>>>(v_w1, wt_v1, 128, 64, 128);
    k_tpad<<<dim3((48 * 64  + 255) / 256), dim3(256), 0, stream>>>(v_w2, wt_v2, 64, 48, 64);
    k_tpad<<<dim3((64 * 128 + 255) / 256), dim3(256), 0, stream>>>(u_w1, wt_u1, 128, 64, 128);
    k_tpad<<<dim3((32 * 64  + 255) / 256), dim3(256), 0, stream>>>(u_w2, wt_u2, 64, 32, 64);

    // 3) edge model (fused LN + scatter)
    int etiles = (E + 15) / 16;
    k_edge<<<dim3((etiles + NW - 1) / NW), dim3(NW * 32), 0, stream>>>(
        x, ea, u, eidx, egid, wt_e1, e_b1, wt_e2, e_b2, e_g, e_be,
        edge_out, agg_e, gsum_e, gcnt_e, E);

    // 4) node model
    int ntiles = (N + 15) / 16;
    k_node<<<dim3((ntiles + NW - 1) / NW), dim3(NW * 32), 0, stream>>>(
        x, agg_e, u, vgid, wt_v1, v_b1, wt_v2, v_b2, v_g, v_be,
        x_out, gsum_v, gcnt_v, N);

    // 5) global model
    int gtiles = (G + 15) / 16;
    k_global<<<dim3((gtiles + NW - 1) / NW), dim3(NW * 32), 0, stream>>>(
        u, gsum_v, gcnt_v, gsum_e, gcnt_e, wt_u1, u_b1, wt_u2, u_b2, u_g, u_be,
        u_out, G);
}
